// HierarchicalMambaCell_82429012344979
// MI455X (gfx1250) — compile-verified, matrix operand order checked by reading, not run on hardware
//
#include <hip/hip_runtime.h>
#include <hip/hip_bf16.h>
#include <cstdint>

typedef _Float16 half_t;
typedef __attribute__((ext_vector_type(16))) _Float16 v16h;
typedef __attribute__((ext_vector_type(8)))  _Float16 v8h;
typedef __attribute__((ext_vector_type(8)))  float    v8f;

#define NTHREADS 256
#define OUTC 726
#define BSZ 8
#define LSEQ 512

// ---------------------------------------------------------------------------
// Block parameter bundle (host-filled, passed by value)
// ---------------------------------------------------------------------------
struct BP {
  int d, ed, r, n;
  int KtIn, NtIn;      // in_proj  GEMM: K = d (pad32), N = 2*ed (pad16)
  int KtX,  NtX;       // x_proj   GEMM: K = ed,       N = r + 2n
  int NtDt;            // dt_w     GEMM: K = r (pad32->1 tile), N = ed
  int KtOut, NtOut;    // out_proj GEMM: K = ed,       N = d
  const half_t *Win, *Wx, *Wdt, *Wout;   // swizzled f16 weights (workspace)
  const float  *negA;                    // -exp(A_log), ed*n
  const float  *convW, *convB, *dtB, *Dv, *normW;  // raw f32 params
  const float  *x_in;       // (B,L,d) input, or nullptr for u-block
  const float  *mean_in;    // means buffer (u-block reads), else nullptr
  float *h;                 // B*ed*n state (workspace)
  float *conv_state;        // B*ed*3 (workspace)
  float *mean_out;          // means buffer (k/v/p write), else nullptr
  int mean_slot;            // 0=k,1=v,2=p
  int out_col;              // column base in 726-wide output
};

// ---------------------------------------------------------------------------
// WMMA GEMM: C(8 x N) = A16(16 x K) * Wsw(K x N); B=8 real rows, rows 8..15 zero.
// A in LDS (row stride 512 halves); W pre-swizzled per 16x32 tile (512 halves,
// lane-major, halves follow the 16-bit operand K-pattern, sel=(lane>=16)*8).
// Templated on Ktiles so the A fragments are loaded from LDS ONCE into
// registers and reused across all N tiles (weight tiles stream from L2).
// ---------------------------------------------------------------------------
template <int KT>
__device__ inline void gemm_wmma_t(const half_t* __restrict__ sA,
                                   const half_t* __restrict__ Wsw,
                                   float* __restrict__ sC, int Cstride,
                                   int Ntiles, int wave, int lane) {
  const int m   = lane & 15;
  const int sel = (lane >> 4) << 3;           // 0 or 8
  const half_t* arow = sA + m * 512;
  v16h a[KT];
#pragma unroll
  for (int kt = 0; kt < KT; ++kt) {
    const half_t* ap = arow + kt * 32 + sel;
    const v8h lo = *(const v8h*)(ap);
    const v8h hi = *(const v8h*)(ap + 16);
#pragma unroll
    for (int i = 0; i < 8; ++i) { a[kt][i] = lo[i]; a[kt][i + 8] = hi[i]; }
  }
  for (int nt = wave; nt < Ntiles; nt += 8) {
    v8f acc = {};
    const half_t* wt = Wsw + (size_t)nt * KT * 512 + lane * 16;
#pragma unroll
    for (int kt = 0; kt < KT; ++kt) {
      const v16h b = *(const v16h*)(wt + (size_t)kt * 512);
      acc = __builtin_amdgcn_wmma_f32_16x16x32_f16(
          /*neg_a=*/false, a[kt], /*neg_b=*/false, b,
          /*c_mod=*/(short)0, acc, /*reuse_a=*/false, /*reuse_b=*/false);
    }
    if (lane < 16) {
#pragma unroll
      for (int v = 0; v < 8; ++v)             // lanes 0..15 hold M=0..7 (batch)
        sC[v * Cstride + nt * 16 + lane] = acc[v];
    }
  }
}

// Generic fallback (A reloaded per tile) for unexpected Ktiles values.
__device__ inline void gemm_wmma_gen(const half_t* __restrict__ sA,
                                     const half_t* __restrict__ Wsw,
                                     float* __restrict__ sC, int Cstride,
                                     int Ktiles, int Ntiles, int wave, int lane) {
  const int m   = lane & 15;
  const int sel = (lane >> 4) << 3;
  const half_t* arow = sA + m * 512;
  for (int nt = wave; nt < Ntiles; nt += 8) {
    v8f acc = {};
    const half_t* wt = Wsw + (size_t)nt * Ktiles * 512 + lane * 16;
    for (int kt = 0; kt < Ktiles; ++kt) {
      const half_t* ap = arow + kt * 32 + sel;
      const v8h lo = *(const v8h*)(ap);
      const v8h hi = *(const v8h*)(ap + 16);
      v16h a;
#pragma unroll
      for (int i = 0; i < 8; ++i) { a[i] = lo[i]; a[i + 8] = hi[i]; }
      const v16h b = *(const v16h*)(wt + (size_t)kt * 512);
      acc = __builtin_amdgcn_wmma_f32_16x16x32_f16(false, a, false, b,
                                                   (short)0, acc, false, false);
    }
    if (lane < 16) {
#pragma unroll
      for (int v = 0; v < 8; ++v)
        sC[v * Cstride + nt * 16 + lane] = acc[v];
    }
  }
}

// Dispatch over the statically known Ktiles values of this model.
__device__ inline void gemm_wmma(const half_t* sA, const half_t* Wsw,
                                 float* sC, int Cstride,
                                 int Ktiles, int Ntiles, int wave, int lane) {
  switch (Ktiles) {
    case 1:  gemm_wmma_t<1>(sA, Wsw, sC, Cstride, Ntiles, wave, lane); break;
    case 2:  gemm_wmma_t<2>(sA, Wsw, sC, Cstride, Ntiles, wave, lane); break;
    case 4:  gemm_wmma_t<4>(sA, Wsw, sC, Cstride, Ntiles, wave, lane); break;
    case 5:  gemm_wmma_t<5>(sA, Wsw, sC, Cstride, Ntiles, wave, lane); break;
    case 8:  gemm_wmma_t<8>(sA, Wsw, sC, Cstride, Ntiles, wave, lane); break;
    case 10: gemm_wmma_t<10>(sA, Wsw, sC, Cstride, Ntiles, wave, lane); break;
    case 16: gemm_wmma_t<16>(sA, Wsw, sC, Cstride, Ntiles, wave, lane); break;
    default: gemm_wmma_gen(sA, Wsw, sC, Cstride, Ktiles, Ntiles, wave, lane);
  }
}

__device__ inline float silu_f(float x) { return x / (1.f + expf(-x)); }

// ---------------------------------------------------------------------------
// Persistent per-block Mamba scan. gridDim 3 (k,v,p) or 1 (u).
// ---------------------------------------------------------------------------
__global__ __launch_bounds__(NTHREADS, 1)
void mamba_scan(BP P0, BP P1, BP P2, float* __restrict__ out) {
  __shared__ float  sXZ[8 * 1024];              // 32 KB  xz = in_proj output
  __shared__ float  sXs[8 * 512];               // 16 KB  conv+silu output
  __shared__ float  sDbc[8 * 128];              //  4 KB  x_proj output
  __shared__ float  sDelta[8 * 512];            // 16 KB
  __shared__ float  sY[8 * 512];                // 16 KB  y * silu(z)
  __shared__ float  sOut[8 * 256];              //  8 KB
  __shared__ float  sX[8 * 256];                //  8 KB  step input (residual)
  __shared__ float  sMean[8 * 64];              //  2 KB
  __shared__ __align__(32) half_t sA[16 * 512]; // 16 KB  f16 A operand staging

  const BP P = (blockIdx.x == 0) ? P0 : ((blockIdx.x == 1) ? P1 : P2);
  const int tid  = threadIdx.x;
  const int lane = tid & 31;
  const int wave = tid >> 5;

  // rows 8..15 of A stay zero forever (pad B=8 -> M=16)
  for (int i = tid; i < 8 * 512; i += NTHREADS) sA[8 * 512 + i] = (half_t)0.f;
  __syncthreads();

  const int d = P.d, ed = P.ed, r = P.r, n = P.n;
  const float inv_d  = 1.0f / (float)d;
  const float inv_ed = 1.0f / (float)ed;

  for (int t = 0; t < LSEQ; ++t) {
    // ---- Stage 0: load step input ------------------------------------
    if (P.x_in) {
      for (int i = tid; i < 8 * d; i += NTHREADS) {
        int b = i / d, c = i % d;
        sX[b * 256 + c] = P.x_in[((size_t)b * LSEQ + t) * d + c];
      }
    } else {  // u-block: concat(mean_k, mean_v, mean_p) at slot t
      for (int i = tid; i < 8 * d; i += NTHREADS) {
        int b = i / d, c = i % d;
        sX[b * 256 + c] =
            P.mean_in[(size_t)t * 1200 + (c / 50) * 400 + b * 50 + (c % 50)];
      }
    }
    __syncthreads();

    // ---- Stage 1: RMSNorm -> f16 A tile (wave b handles row b) -------
    {
      int b = wave;
      float ss = 0.f;
      for (int c = lane; c < d; c += 32) { float v = sX[b * 256 + c]; ss += v * v; }
#pragma unroll
      for (int off = 16; off > 0; off >>= 1) ss += __shfl_xor(ss, off, 32);
      float scale = rsqrtf(ss * inv_d + 1e-5f);
      int Kpad = P.KtIn * 32;
      for (int c = lane; c < Kpad; c += 32) {
        float v = (c < d) ? sX[b * 256 + c] * scale * P.normW[c] : 0.f;
        sA[b * 512 + c] = (half_t)v;
      }
    }
    __syncthreads();

    // ---- Stage 2: xz = xn @ in_proj^T --------------------------------
    gemm_wmma(sA, P.Win, sXZ, 1024, P.KtIn, P.NtIn, wave, lane);
    __syncthreads();

    // ---- Stage 3: depthwise conv (width 4) + silu --------------------
    for (int i = tid; i < 8 * ed; i += NTHREADS) {
      int b = i / ed, e = i % ed;
      float xi = sXZ[b * 1024 + e];
      float* cs = P.conv_state + ((size_t)b * ed + e) * 3;
      float c0 = cs[0], c1 = cs[1], c2 = cs[2];
      const float* cw = P.convW + e * 4;
      float xc = c0 * cw[0] + c1 * cw[1] + c2 * cw[2] + xi * cw[3] + P.convB[e];
      sXs[b * 512 + e] = silu_f(xc);
      cs[0] = c1; cs[1] = c2; cs[2] = xi;
    }
    __syncthreads();

    // ---- Stage 4: stage xs as f16 A ----------------------------------
    {
      int Kpad = P.KtX * 32;
      for (int i = tid; i < 8 * Kpad; i += NTHREADS) {
        int b = i / Kpad, c = i % Kpad;
        sA[b * 512 + c] = (half_t)((c < ed) ? sXs[b * 512 + c] : 0.f);
      }
    }
    __syncthreads();

    // ---- Stage 5: dbc = xs @ x_proj^T --------------------------------
    gemm_wmma(sA, P.Wx, sDbc, 128, P.KtX, P.NtX, wave, lane);
    __syncthreads();

    // ---- Stage 6: stage dbc[:, :r] as f16 A (single K tile) ----------
    for (int i = tid; i < 8 * 32; i += NTHREADS) {
      int b = i / 32, c = i % 32;
      sA[b * 512 + c] = (half_t)((c < r) ? sDbc[b * 128 + c] : 0.f);
    }
    __syncthreads();

    // ---- Stage 7: delta_raw = dbc_r @ dt_w^T -------------------------
    gemm_wmma(sA, P.Wdt, sDelta, 512, 1, P.NtDt, wave, lane);
    __syncthreads();

    // ---- Stage 8: softplus(delta + dt_b); zero mean accumulator ------
    for (int i = tid; i < 8 * ed; i += NTHREADS) {
      int b = i / ed, e = i % ed;
      float x = sDelta[b * 512 + e] + P.dtB[e];
      sDelta[b * 512 + e] = (x > 20.f) ? x : log1pf(expf(x));
    }
    for (int i = tid; i < 8 * 64; i += NTHREADS) sMean[i] = 0.f;
    __syncthreads();

    // ---- Stage 9: SSM state update + y, fused with silu(z) -----------
    {
      const bool domean = (P.mean_out != nullptr);
      for (int i = tid; i < 8 * ed; i += NTHREADS) {
        int b = i / ed, e = i % ed;
        float delta = sDelta[b * 512 + e];
        float xs    = sXs[b * 512 + e];
        float dx    = delta * xs;
        float* hp        = P.h    + ((size_t)b * ed + e) * n;
        const float* Ae  = P.negA + (size_t)e * n;
        const float* Bm  = &sDbc[b * 128 + r];
        const float* Cm  = &sDbc[b * 128 + r + n];
        float yacc = 0.f;
        for (int j = 0; j < n; ++j) {
          float h2 = expf(delta * Ae[j]) * hp[j] + dx * Bm[j];
          hp[j] = h2;
          yacc += h2 * Cm[j];
          if (domean) atomicAdd(&sMean[b * 64 + j], h2 * inv_ed);
        }
        float y = yacc + P.Dv[e] * xs;
        float z = sXZ[b * 1024 + ed + e];
        sY[b * 512 + e] = y * silu_f(z);
      }
    }
    __syncthreads();

    // ---- Stage 10: stage y*silu(z) as f16 A --------------------------
    {
      int Kpad = P.KtOut * 32;
      for (int i = tid; i < 8 * Kpad; i += NTHREADS) {
        int b = i / Kpad, c = i % Kpad;
        sA[b * 512 + c] = (half_t)((c < ed) ? sY[b * 512 + c] : 0.f);
      }
    }
    __syncthreads();

    // ---- Stage 11: out = yz @ out_proj^T -----------------------------
    gemm_wmma(sA, P.Wout, sOut, 256, P.KtOut, P.NtOut, wave, lane);
    __syncthreads();

    // ---- Stage 12: residual + store; publish state means -------------
    for (int i = tid; i < 8 * d; i += NTHREADS) {
      int b = i / d, c = i % d;
      out[((size_t)b * LSEQ + t) * OUTC + P.out_col + c] =
          sOut[b * 256 + c] + sX[b * 256 + c];
    }
    if (P.mean_out) {
      for (int i = tid; i < 8 * n; i += NTHREADS) {
        int b = i / n, j = i % n;
        P.mean_out[(size_t)(t + 1) * 1200 + P.mean_slot * 400 + b * 50 + j] =
            sMean[b * 64 + j];
      }
    }
    __syncthreads();
  }
}

// ---------------------------------------------------------------------------
// Prep kernels
// ---------------------------------------------------------------------------
// Swizzle f32 weight W (Ntot x Ktot, row-major, used as W^T B-operand) into
// per-tile f16 layout consumed by gemm_wmma (zero-padded).
__global__ void swizzle_w(const float* __restrict__ W, half_t* __restrict__ dst,
                          int Ntot, int Ktot, int Ntiles, int Ktiles) {
  int total = Ntiles * Ktiles * 512;
  for (int idx = blockIdx.x * blockDim.x + threadIdx.x; idx < total;
       idx += gridDim.x * blockDim.x) {
    int h    = idx & 15;
    int lane = (idx >> 4) & 31;
    int kt   = (idx >> 9) % Ktiles;
    int nt   = idx / (512 * Ktiles);
    int nn   = nt * 16 + (lane & 15);
    int sel  = (lane >> 4) << 3;
    int k    = kt * 32 + ((h < 8) ? (sel + h) : (16 + sel + (h - 8)));
    float v  = (nn < Ntot && k < Ktot) ? W[(size_t)nn * Ktot + k] : 0.f;
    dst[idx] = (half_t)v;
  }
}

__global__ void neg_exp_k(const float* __restrict__ s, float* __restrict__ dst, int nelem) {
  for (int i = blockIdx.x * blockDim.x + threadIdx.x; i < nelem;
       i += gridDim.x * blockDim.x)
    dst[i] = -expf(s[i]);
}

__global__ void zero_k(float* __restrict__ dst, int nelem) {
  for (int i = blockIdx.x * blockDim.x + threadIdx.x; i < nelem;
       i += gridDim.x * blockDim.x)
    dst[i] = 0.f;
}

// ---------------------------------------------------------------------------
// Host launch
// ---------------------------------------------------------------------------
extern "C" void kernel_launch(void* const* d_in, const int* in_sizes, int n_in,
                              void* d_out, int out_size, void* d_ws, size_t ws_size,
                              hipStream_t stream) {
  (void)in_sizes; (void)n_in; (void)out_size; (void)ws_size;

  // input order: xf, xk, xv, xp, then params k/v/p/u each:
  // in_proj, conv_w, conv_b, x_proj, dt_w, dt_b, A_log, D, out_proj, norm_w
  const float* x_in[3] = { (const float*)d_in[1],   // xk
                           (const float*)d_in[2],   // xv
                           (const float*)d_in[3] }; // xp
  const int pbase[4] = {4, 14, 24, 34};
  struct Dims { int d, ed, r, n; };
  const Dims dims[4] = { {256, 512, 16, 50}, {256, 512, 16, 50},
                         {64, 128, 4, 50},   {150, 300, 10, 20} };
  const int out_cols[4] = {0, 256, 512, 576};

  size_t off = 0;
  auto alloc = [&](size_t bytes) -> char* {
    size_t a = (off + 255) & ~(size_t)255;
    off = a + bytes;
    return (char*)d_ws + a;
  };
  auto grid_for = [](int total) {
    int g = (total + NTHREADS - 1) / NTHREADS;
    return (g > 2048) ? 2048 : (g < 1 ? 1 : g);
  };
  auto launch_swz = [&](const float* W, half_t* dst, int Ntot, int Ktot,
                        int Ntiles, int Ktiles) {
    int total = Ntiles * Ktiles * 512;
    swizzle_w<<<grid_for(total), NTHREADS, 0, stream>>>(W, dst, Ntot, Ktot,
                                                        Ntiles, Ktiles);
  };

  BP bp[4];
  for (int blk = 0; blk < 4; ++blk) {
    const Dims& D = dims[blk];
    const int b0 = pbase[blk];
    const float* in_proj  = (const float*)d_in[b0 + 0];
    const float* conv_w   = (const float*)d_in[b0 + 1];
    const float* conv_b   = (const float*)d_in[b0 + 2];
    const float* x_proj   = (const float*)d_in[b0 + 3];
    const float* dt_w     = (const float*)d_in[b0 + 4];
    const float* dt_b     = (const float*)d_in[b0 + 5];
    const float* A_log    = (const float*)d_in[b0 + 6];
    const float* Dvec     = (const float*)d_in[b0 + 7];
    const float* out_proj = (const float*)d_in[b0 + 8];
    const float* norm_w   = (const float*)d_in[b0 + 9];

    BP& P = bp[blk];
    P.d = D.d; P.ed = D.ed; P.r = D.r; P.n = D.n;
    P.KtIn  = (D.d + 31) / 32;       P.NtIn  = (2 * D.ed + 15) / 16;
    P.KtX   = (D.ed + 31) / 32;      P.NtX   = (D.r + 2 * D.n + 15) / 16;
    P.NtDt  = (D.ed + 15) / 16;
    P.KtOut = P.KtX;                 P.NtOut = (D.d + 15) / 16;

    half_t* win  = (half_t*)alloc((size_t)P.NtIn  * P.KtIn * 512 * 2);
    half_t* wx   = (half_t*)alloc((size_t)P.NtX   * P.KtX  * 512 * 2);
    half_t* wdt  = (half_t*)alloc((size_t)P.NtDt  * 1      * 512 * 2);
    half_t* wout = (half_t*)alloc((size_t)P.NtOut * P.KtOut* 512 * 2);
    launch_swz(in_proj,  win,  2 * D.ed,        D.d,  P.NtIn,  P.KtIn);
    launch_swz(x_proj,   wx,   D.r + 2 * D.n,   D.ed, P.NtX,   P.KtX);
    launch_swz(dt_w,     wdt,  D.ed,            D.r,  P.NtDt,  1);
    launch_swz(out_proj, wout, D.d,             D.ed, P.NtOut, P.KtOut);
    P.Win = win; P.Wx = wx; P.Wdt = wdt; P.Wout = wout;

    float* negA = (float*)alloc((size_t)D.ed * D.n * 4);
    neg_exp_k<<<grid_for(D.ed * D.n), NTHREADS, 0, stream>>>(A_log, negA,
                                                             D.ed * D.n);
    P.negA = negA;

    float* h  = (float*)alloc((size_t)BSZ * D.ed * D.n * 4);
    float* cs = (float*)alloc((size_t)BSZ * D.ed * 3 * 4);
    zero_k<<<grid_for(BSZ * D.ed * D.n), NTHREADS, 0, stream>>>(h, BSZ * D.ed * D.n);
    zero_k<<<grid_for(BSZ * D.ed * 3), NTHREADS, 0, stream>>>(cs, BSZ * D.ed * 3);
    P.h = h; P.conv_state = cs;

    P.convW = conv_w; P.convB = conv_b; P.dtB = dt_b; P.Dv = Dvec; P.normW = norm_w;
    P.x_in = (blk < 3) ? x_in[blk] : nullptr;
    P.mean_in = nullptr; P.mean_out = nullptr;
    P.mean_slot = (blk < 3) ? blk : 0;
    P.out_col = out_cols[blk];
  }

  // means buffer: [L+1][3 blocks][B=8][50]; slot 0 = zeros (initial carry)
  float* means = (float*)alloc((size_t)(LSEQ + 1) * 1200 * 4);
  zero_k<<<grid_for(1200), NTHREADS, 0, stream>>>(means, 1200);
  bp[0].mean_out = means; bp[1].mean_out = means; bp[2].mean_out = means;
  bp[3].mean_in = means;

  float* out = (float*)d_out;
  // Phase 1: k, v, p scans (independent, one WGP each)
  mamba_scan<<<3, NTHREADS, 0, stream>>>(bp[0], bp[1], bp[2], out);
  // Phase 2: u scan (reads per-step means written by phase 1)
  mamba_scan<<<1, NTHREADS, 0, stream>>>(bp[3], bp[3], bp[3], out);
}
